// GCLSTM_45638322487636
// MI455X (gfx1250) — compile-verified, matrix operand order checked by reading, not run on hardware
//
#include <hip/hip_runtime.h>
#include <hip/hip_bf16.h>

#define NNODES 100000
#define NEDGES 1600000
#define DDIM   128

typedef __attribute__((ext_vector_type(16))) __bf16 v16bf;
typedef __attribute__((ext_vector_type(8)))  __bf16 v8bf;
typedef __attribute__((ext_vector_type(8)))  float  v8f;

__device__ __forceinline__ unsigned short f2bf(float f) {
    unsigned u = __builtin_bit_cast(unsigned, f);
    u += 0x7FFFu + ((u >> 16) & 1u);          // round-to-nearest-even
    return (unsigned short)(u >> 16);
}

__device__ __forceinline__ float sigm(float x) { return 1.0f / (1.0f + __expf(-x)); }

// ---------------- graph preprocessing ----------------

__global__ void k_zero(float* __restrict__ p, int n) {
    int i = blockIdx.x * blockDim.x + threadIdx.x;
    if (i < n) p[i] = 0.0f;
}

__global__ void k_neg(const float* __restrict__ s, float* __restrict__ d, int n) {
    int i = blockIdx.x * blockDim.x + threadIdx.x;
    if (i < n) d[i] = -s[i];
}

__global__ void k_deg(const int* __restrict__ ei, const float* __restrict__ ew,
                      float* __restrict__ deg) {
    int e = blockIdx.x * blockDim.x + threadIdx.x;
    if (e < NEDGES) atomicAdd(&deg[ei[e]], ew[e]);
}

__global__ void k_dinv(const float* __restrict__ deg, float* __restrict__ dinv) {
    int i = blockIdx.x * blockDim.x + threadIdx.x;
    if (i < NNODES) {
        float d = deg[i];
        dinv[i] = d > 0.0f ? rsqrtf(d) : 0.0f;
    }
}

__global__ void k_wedge(const int* __restrict__ ei, const float* __restrict__ ew,
                        const float* __restrict__ dinv, float* __restrict__ wE) {
    int e = blockIdx.x * blockDim.x + threadIdx.x;
    if (e < NEDGES) {
        int r = ei[e], c = ei[NEDGES + e];
        wE[e] = -(dinv[r] * ew[e] * dinv[c]);   // * (2/lambda_max) == 1
    }
}

// dst[col] += scale * wE[e] * src[row]   (one wave per edge; lane covers 4 cols)
__global__ void k_spmm(const int* __restrict__ ei, const float* __restrict__ wE,
                       const float* __restrict__ src, float* __restrict__ dst,
                       float scale) {
    int g = blockIdx.x * blockDim.x + threadIdx.x;
    int e = g >> 5, lane = g & 31;
    if (e >= NEDGES) return;
    float w = scale * wE[e];
    int r = ei[e], c = ei[NEDGES + e];
    const float4 v = *(const float4*)(src + (size_t)r * DDIM + lane * 4);
    float* d = dst + (size_t)c * DDIM + lane * 4;
    atomicAdd(d + 0, w * v.x);
    atomicAdd(d + 1, w * v.y);
    atomicAdd(d + 2, w * v.z);
    atomicAdd(d + 3, w * v.w);
}

// ---------------- weight repack: Wbig(512x512) -> bf16 WMMA-fragment-major ----
// Wp[((kt*32 + ct)*32 + lane)*16 + j]; lane = n + 16*half; k_local follows the
// ISA 16-bit A/B fragment layout (pairs 0-3 -> K base, pairs 4-7 -> K base+16).

__global__ void k_pack(const float* __restrict__ W, const float* __restrict__ cW,
                       unsigned short* __restrict__ Wp) {
    int idx = blockIdx.x * blockDim.x + threadIdx.x;   // 262144 total
    int j    = idx & 15;
    int lane = (idx >> 4) & 31;
    int ct   = (idx >> 9) & 31;
    int kt   = idx >> 14;
    int p = j >> 1, q = j & 1;
    int half = lane >> 4, n = lane & 15;
    int klocal = ((p >> 2) << 4) + (half << 3) + ((p & 3) << 1) + q;
    int k = kt * 32 + klocal;                 // concat-input dim (0..511)
    int c = ct * 16 + n;                      // output dim (gate*128 + d)
    int g = c >> 7, d = c & 127;
    float v;
    if (k < 128)      v = W [(g * 128           + k        ) * 128 + d];
    else if (k < 256) v = cW[((g * 3 + 0) * 128 + (k - 128)) * 128 + d];
    else if (k < 384) v = cW[((g * 3 + 1) * 128 + (k - 256)) * 128 + d];
    else              v = cW[((g * 3 + 2) * 128 + (k - 384)) * 128 + d];
    Wp[idx] = f2bf(v);
}

// ---------------- fused GEMM (bf16 WMMA, f32 acc) + LSTM epilogue ------------
// block = 128 threads = 4 waves; 32-node row tile (2 16-row WMMA tiles / wave).
// Wave w owns output columns d in [w*32, w*32+32) replicated across ALL 4
// gates (column tiles ct = g*8 + w*2 + s), so the LSTM mixing i/f/c/o happens
// entirely inside the wave's own accumulators -- no LDS round trip.

__global__ void __launch_bounds__(128)
k_gemm_lstm(const float* __restrict__ X, const float* __restrict__ H,
            const float* __restrict__ T1, const float* __restrict__ T2,
            const unsigned short* __restrict__ Wp,
            const float* __restrict__ b, const float* __restrict__ cb,
            const float* __restrict__ Cst, float* __restrict__ out) {
    __shared__ __align__(16) unsigned short lds_a[32 * 520]; // 32x512 bf16, +8 pad

    const int tid  = threadIdx.x;
    const int wave = tid >> 5;
    const int lane = tid & 31;
    const long nbase = (long)blockIdx.x * 32;

    // stage A tile (32 rows x 512 k) as bf16 into LDS; float4 loads, b64 stores
    for (int i = 0; i < 32; ++i) {
        int idx4 = tid + (i << 7);            // 0..4095 float4 elements
        int m  = idx4 >> 7;                   // row 0..31
        int k4 = idx4 & 127;                  // float4 index within 512-k row
        int k  = k4 << 2;
        long node = nbase + m;
        float4 v;
        if (k < 128)      v = *(const float4*)(X  + node * DDIM + k);
        else if (k < 256) v = *(const float4*)(H  + node * DDIM + (k - 128));
        else if (k < 384) v = *(const float4*)(T1 + node * DDIM + (k - 256));
        else              v = *(const float4*)(T2 + node * DDIM + (k - 384));
        unsigned lo = (unsigned)f2bf(v.x) | ((unsigned)f2bf(v.y) << 16);
        unsigned hi = (unsigned)f2bf(v.z) | ((unsigned)f2bf(v.w) << 16);
        uint2* dst = (uint2*)&lds_a[m * 520 + k];
        *dst = make_uint2(lo, hi);
    }
    __syncthreads();

    v8f acc[2][8] = {};                       // [row tile][gate*2 + s]
    const int n    = lane & 15;
    const int half = lane >> 4;
    const v16bf* Bp = (const v16bf*)Wp;

    for (int kt = 0; kt < 16; ++kt) {
        // A fragments for the two 16-row tiles (ISA 16-bit A layout)
        const unsigned short* ap0 = &lds_a[(n)      * 520 + kt * 32 + half * 8];
        const unsigned short* ap1 = &lds_a[(n + 16) * 520 + kt * 32 + half * 8];
        v8bf a0lo = *(const v8bf*)(ap0);
        v8bf a0hi = *(const v8bf*)(ap0 + 16);
        v8bf a1lo = *(const v8bf*)(ap1);
        v8bf a1hi = *(const v8bf*)(ap1 + 16);
        v16bf a0 = __builtin_shufflevector(a0lo, a0hi,
                    0, 1, 2, 3, 4, 5, 6, 7, 8, 9, 10, 11, 12, 13, 14, 15);
        v16bf a1 = __builtin_shufflevector(a1lo, a1hi,
                    0, 1, 2, 3, 4, 5, 6, 7, 8, 9, 10, 11, 12, 13, 14, 15);
#pragma unroll
        for (int g = 0; g < 4; ++g) {
#pragma unroll
            for (int s = 0; s < 2; ++s) {
                int ct = g * 8 + wave * 2 + s;
                v16bf bb = Bp[(kt * 32 + ct) * 32 + lane];
                acc[0][g * 2 + s] = __builtin_amdgcn_wmma_f32_16x16x32_bf16(
                    false, a0, false, bb, (short)0, acc[0][g * 2 + s], false, false);
                acc[1][g * 2 + s] = __builtin_amdgcn_wmma_f32_16x16x32_bf16(
                    false, a1, false, bb, (short)0, acc[1][g * 2 + s], false, false);
            }
        }
    }

    // LSTM epilogue straight from accumulators.
    // C/D layout: VGPR r -> M = r + 8*half, N = lane&15.
#pragma unroll
    for (int s = 0; s < 2; ++s) {
        int colg = (wave * 2 + s) * 16 + n;   // column within gate, 0..127
        float bi = b[0   + colg] + cb[0   + colg];
        float bf = b[128 + colg] + cb[128 + colg];
        float bc = b[256 + colg] + cb[256 + colg];
        float bo = b[384 + colg] + cb[384 + colg];
#pragma unroll
        for (int rt = 0; rt < 2; ++rt) {
#pragma unroll
            for (int r = 0; r < 8; ++r) {
                long node = nbase + rt * 16 + r + 8 * half;
                float gi = acc[rt][0 + s][r] + bi;
                float gf = acc[rt][2 + s][r] + bf;
                float gc = acc[rt][4 + s][r] + bc;
                float go = acc[rt][6 + s][r] + bo;
                float I = sigm(gi), F = sigm(gf);
                float T = tanhf(gc), O = sigm(go);
                float cn = F * Cst[node * DDIM + colg] + I * T;
                out[node * DDIM + colg] = O * tanhf(cn);
            }
        }
    }
}

// ---------------- launch ----------------

extern "C" void kernel_launch(void* const* d_in, const int* in_sizes, int n_in,
                              void* d_out, int out_size, void* d_ws, size_t ws_size,
                              hipStream_t stream) {
    (void)in_sizes; (void)n_in; (void)out_size; (void)ws_size;
    const float* X  = (const float*)d_in[0];
    const int*   ei = (const int*)  d_in[1];
    const float* ew = (const float*)d_in[2];
    const float* H  = (const float*)d_in[3];
    const float* C  = (const float*)d_in[4];
    const float* W  = (const float*)d_in[5];
    const float* b  = (const float*)d_in[6];
    const float* cW = (const float*)d_in[7];
    const float* cb = (const float*)d_in[8];
    float* out = (float*)d_out;

    char* ws = (char*)d_ws;
    size_t off = 0;
    auto alloc = [&](size_t bytes) -> void* {
        void* p = ws + off;
        off = (off + bytes + 255) & ~(size_t)255;
        return p;
    };
    float* deg  = (float*)alloc((size_t)NNODES * 4);
    float* dinv = (float*)alloc((size_t)NNODES * 4);
    float* wE   = (float*)alloc((size_t)NEDGES * 4);
    float* T1   = (float*)alloc((size_t)NNODES * DDIM * 4);
    float* T2   = (float*)alloc((size_t)NNODES * DDIM * 4);
    unsigned short* Wp = (unsigned short*)alloc((size_t)512 * 512 * 2);

    const int ND = NNODES * DDIM;
    k_zero <<<(NNODES + 255) / 256, 256, 0, stream>>>(deg, NNODES);
    k_deg  <<<(NEDGES + 255) / 256, 256, 0, stream>>>(ei, ew, deg);
    k_dinv <<<(NNODES + 255) / 256, 256, 0, stream>>>(deg, dinv);
    k_wedge<<<(NEDGES + 255) / 256, 256, 0, stream>>>(ei, ew, dinv, wE);
    k_zero <<<(ND + 255) / 256, 256, 0, stream>>>(T1, ND);
    k_spmm <<<NEDGES / 8, 256, 0, stream>>>(ei, wE, H, T1, 1.0f);   // Tx1 = S H
    k_neg  <<<(ND + 255) / 256, 256, 0, stream>>>(H, T2, ND);       // Tx2 = -H
    k_spmm <<<NEDGES / 8, 256, 0, stream>>>(ei, wE, T1, T2, 2.0f);  // Tx2 += 2 S Tx1
    k_pack <<<(512 * 512) / 256, 256, 0, stream>>>(W, cW, Wp);
    k_gemm_lstm<<<NNODES / 32, 128, 0, stream>>>(X, H, T1, T2, Wp, b, cb, C, out);
}
